// AFM_83348135346298
// MI455X (gfx1250) — matrix operand order repaired and checked by reading, not compiled, not run on hardware
//
#include <hip/hip_runtime.h>
#include <hip/hip_bf16.h>

typedef _Float16 half_t;
typedef __attribute__((ext_vector_type(2)))  _Float16 v2h;
typedef __attribute__((ext_vector_type(4)))  _Float16 v4h;
typedef __attribute__((ext_vector_type(16))) _Float16 v16h;
typedef __attribute__((ext_vector_type(8)))  float    v8f;

#define VALID_DIM 20
#define NPAIR     190          // 20*19/2
#define PAD_PAIR  192          // padded to 12 WMMA M-tiles
#define EMB_DIM   64
#define ATT_DIM   32
#define SCALE     512.0f       // keep f16 ele values out of subnormal range
#define INV_SCALE (1.0f / 512.0f)

__launch_bounds__(128)
__global__ void afm_fused_kernel(
    const int*   __restrict__ features,
    const float* __restrict__ feature_values,
    const float* __restrict__ emb_table,
    const float* __restrict__ bias_table,
    const float* __restrict__ global_bias,
    const float* __restrict__ att_w1,
    const float* __restrict__ att_b1,
    const float* __restrict__ att_w2,
    const float* __restrict__ pred_w,
    float*       __restrict__ out)
{
    __shared__ int    s_feat[VALID_DIM];
    __shared__ float  s_fv[VALID_DIM];
    __shared__ __align__(16) float  s_e[VALID_DIM * EMB_DIM];     //  5 KB
    __shared__ __align__(16) half_t s_w1[ATT_DIM * EMB_DIM];      //  4 KB
    __shared__ __align__(16) half_t s_ele[PAD_PAIR * EMB_DIM];    // 24 KB
    __shared__ float  s_logits[PAD_PAIR];   // reused as partial scratch later
    __shared__ float  s_att[PAD_PAIR];
    __shared__ float  s_red[2];
    __shared__ float  s_fb;

    const int b    = blockIdx.x;
    const int tid  = threadIdx.x;
    const int lane = tid & 31;        // wave32
    const int wave = tid >> 5;        // 4 waves
    const int r    = lane & 15;
    const int hi   = lane >> 4;

    // ---- stage features/values; linear-bias term via wave-0 shuffle reduce --
    if (tid < VALID_DIM) {
        s_feat[tid] = features[b * VALID_DIM + tid];
        s_fv[tid]   = feature_values[b * VALID_DIM + tid];
    }
    if (wave == 0) {
        float fbv = 0.0f;
        if (lane < VALID_DIM) {
            int f = features[b * VALID_DIM + lane];
            fbv = bias_table[f] * feature_values[b * VALID_DIM + lane];
        }
        #pragma unroll
        for (int m = 16; m >= 1; m >>= 1) fbv += __shfl_xor(fbv, m, 32);
        if (lane == 0) s_fb = fbv;
    }
    // stage W1 as f16 (row-major [ATT_DIM][EMB_DIM], same as input layout)
    for (int idx = tid; idx < ATT_DIM * EMB_DIM; idx += 128)
        s_w1[idx] = (half_t)att_w1[idx];
    // zero the 2 padded pair rows (2*64 = 128 halves, one per thread)
    s_ele[NPAIR * EMB_DIM + tid] = (half_t)0.0f;
    __syncthreads();

    // ---- gather + scale embeddings (float4 vectorized, table is L2-resident)
    for (int q = tid; q < VALID_DIM * (EMB_DIM / 4); q += 128) {
        int f  = q >> 4;
        int d4 = q & 15;
        const float4 v =
            ((const float4*)(emb_table + (size_t)s_feat[f] * EMB_DIM))[d4];
        float s = s_fv[f];
        ((float4*)s_e)[q] = make_float4(v.x * s, v.y * s, v.z * s, v.w * s);
    }
    __syncthreads();

    // ---- pairwise products, scaled into f16 (float4 in, v4h out) -----------
    for (int p = tid; p < NPAIR; p += 128) {
        int i = 0, rem = p, cnt = VALID_DIM - 1;
        while (rem >= cnt) { rem -= cnt; ++i; --cnt; }
        int j = i + 1 + rem;
        const float4* a4 = (const float4*)(s_e + i * EMB_DIM);
        const float4* b4 = (const float4*)(s_e + j * EMB_DIM);
        v4h* dst4 = (v4h*)(s_ele + p * EMB_DIM);
        #pragma unroll
        for (int q = 0; q < EMB_DIM / 4; ++q) {
            float4 a = a4[q], c = b4[q];
            v4h o;
            o.x = (half_t)(a.x * c.x * SCALE);
            o.y = (half_t)(a.y * c.y * SCALE);
            o.z = (half_t)(a.z * c.z * SCALE);
            o.w = (half_t)(a.w * c.w * SCALE);
            dst4[q] = o;
        }
    }
    __syncthreads();

    // ---- attention MLP via WMMA: h = relu((ele_s @ W1^T)/S + b1) ------------
    // logits[p] = sum_a h[p][a] * w2[a]; each wave owns 3 of 12 M-tiles.
    // Defer the 16-lane reduction until both N-tiles are accumulated.
    for (int mt = wave * 3; mt < wave * 3 + 3; ++mt) {
        const int m0 = mt * 16;
        float contrib[8];
        #pragma unroll
        for (int v = 0; v < 8; ++v) contrib[v] = 0.0f;

        #pragma unroll
        for (int nt = 0; nt < 2; ++nt) {
            const int n = nt * 16 + r;
            v8f c = {};                          // zero accum; bias added later
            #pragma unroll
            for (int kt = 0; kt < 2; ++kt) {
                // A fragment (16x32 f16): lane r = row, hi selects K groups
                //   v<4 : K = hi*8 + 2v      ; v>=4 : K = 16 + hi*8 + 2(v-4)
                union { v16h v; unsigned u[8]; } a;
                const unsigned short* arow =
                    (const unsigned short*)(s_ele + (m0 + r) * EMB_DIM + kt * 32);
                #pragma unroll
                for (int vv = 0; vv < 8; ++vv) {
                    int k0 = (vv < 4) ? (hi * 8 + 2 * vv)
                                      : (16 + hi * 8 + 2 * (vv - 4));
                    a.u[vv] = *(const unsigned*)(arow + k0);
                }
                // B fragment (32x16 f16): B[k][n] = W1[n][k]; same K order as A:
                // elements 0-7 hold K = hi*8 + s, elements 8-15 hold
                // K = 16 + hi*8 + (s-8) -> two 8-half runs of the W1 row.
                union { v16h v; uint4 q[2]; } bf;
                const uint4* bp =
                    (const uint4*)(s_w1 + n * EMB_DIM + kt * 32 + hi * 8);
                bf.q[0] = bp[0];
                bf.q[1] = bp[2];
                c = __builtin_amdgcn_wmma_f32_16x16x32_f16(
                        false, a.v, false, bf.v, (short)0, c, false, false);
            }
            const float bias_n = att_b1[n];
            const float w2_n   = att_w2[n];
            #pragma unroll
            for (int v = 0; v < 8; ++v)
                contrib[v] += fmaxf(c[v] * INV_SCALE + bias_n, 0.0f) * w2_n;
        }
        // one 16-lane reduction per row (xor masks 1,2,4,8 stay in-half)
        #pragma unroll
        for (int v = 0; v < 8; ++v) {
            float x = contrib[v];
            #pragma unroll
            for (int m = 8; m >= 1; m >>= 1) x += __shfl_xor(x, m, 32);
            contrib[v] = x;
        }
        if (r == 0) {                             // one writer per row
            #pragma unroll
            for (int v = 0; v < 8; ++v) {
                int m = m0 + v + hi * 8;          // C/D layout: M = v + 8*hi
                if (m < NPAIR) s_logits[m] = contrib[v];
            }
        }
    }
    __syncthreads();

    // ---- softmax over 190 pairs (wave 0, shuffle reductions) ----------------
    if (wave == 0) {
        float mx = -3.0e38f;
        for (int p = lane; p < NPAIR; p += 32) mx = fmaxf(mx, s_logits[p]);
        #pragma unroll
        for (int m = 16; m >= 1; m >>= 1) mx = fmaxf(mx, __shfl_xor(mx, m, 32));
        float sum = 0.0f;
        for (int p = lane; p < NPAIR; p += 32) {
            float ex = __expf(s_logits[p] - mx);
            s_att[p] = ex;
            sum += ex;
        }
        #pragma unroll
        for (int m = 16; m >= 1; m >>= 1) sum += __shfl_xor(sum, m, 32);
        float inv = 1.0f / sum;
        for (int p = lane; p < NPAIR; p += 32) s_att[p] *= inv;
    }
    __syncthreads();

    // ---- afm[d] = sum_p att[p]*ele[p][d]; pred = afm . pred_w ---------------
    // all 128 threads: d = tid&63, pair-half = tid>>6 (95 pairs each)
    {
        const int d    = tid & 63;
        const int phal = tid >> 6;
        const int p0   = phal * 95;
        const half_t* col = s_ele + d;
        float acc = 0.0f;
        for (int p = p0; p < p0 + 95; ++p)
            acc = fmaf(s_att[p], (float)col[p * EMB_DIM], acc);
        s_logits[tid] = acc;                      // reuse s_logits as scratch
    }
    __syncthreads();
    float part = 0.0f;
    if (tid < EMB_DIM)
        part = (s_logits[tid] + s_logits[tid + 64]) * INV_SCALE * pred_w[tid];
    if (wave < 2) {
        #pragma unroll
        for (int m = 16; m >= 1; m >>= 1) part += __shfl_xor(part, m, 32);
        if (lane == 0) s_red[wave] = part;
    }
    __syncthreads();
    if (tid == 0)
        out[b] = s_red[0] + s_red[1] + s_fb + global_bias[0];
}

extern "C" void kernel_launch(void* const* d_in, const int* in_sizes, int n_in,
                              void* d_out, int out_size, void* d_ws, size_t ws_size,
                              hipStream_t stream) {
    const int*   features       = (const int*)  d_in[0];
    const float* feature_values = (const float*)d_in[1];
    const float* emb_table      = (const float*)d_in[2];
    const float* bias_table     = (const float*)d_in[3];
    const float* global_bias    = (const float*)d_in[4];
    const float* att_w1         = (const float*)d_in[5];
    const float* att_b1         = (const float*)d_in[6];
    const float* att_w2         = (const float*)d_in[7];
    const float* pred_w         = (const float*)d_in[8];
    float*       out            = (float*)d_out;

    const int B = in_sizes[0] / VALID_DIM;   // 8192
    afm_fused_kernel<<<B, 128, 0, stream>>>(
        features, feature_values, emb_table, bias_table, global_bias,
        att_w1, att_b1, att_w2, pred_w, out);
}
